// GraphAllEdgeNet_8933531975982
// MI455X (gfx1250) — matrix operand order repaired and checked by reading, not compiled, not run on hardware
//
#include <hip/hip_runtime.h>
#include <math.h>

#define N_NODES 100000
#define N_EDGES 500000
#define CH 64
#define CIN 128
#define TILE 128   // edges per workgroup (8 waves x 16 edges)

typedef __attribute__((ext_vector_type(16))) __bf16 bf16x16;
typedef __attribute__((ext_vector_type(8)))  float  f32x8;

struct BN { const float *g, *b, *m, *v; };
struct ECIn { BN bn1; const float* W1; BN bn2; const float* W2; };

struct PrepIn {
  BN bn0;
  ECIn ec[4];
  BN obn[3];
};
struct PrepOut {
  float *a0, *c0;
  float *a1[4], *c1[4], *a2[4], *c2[4];
  __bf16 *W1bf[4], *W2bf[4];
  float *oa[3], *oc[3];
};

// ---------------- parameter folding: bn -> a*x+c, weights -> bf16 -------------
__device__ inline void fold_bn(const BN& bn, float* a, float* c, int n, int t, int nt) {
  for (int i = t; i < n; i += nt) {
    float aa = bn.g[i] * rsqrtf(bn.v[i] + 1e-5f);
    a[i] = aa;
    c[i] = bn.b[i] - bn.m[i] * aa;
  }
}

__global__ void prep_kernel(PrepIn in, PrepOut out) {
  int t = threadIdx.x, nt = blockDim.x;
  fold_bn(in.bn0, out.a0, out.c0, CH, t, nt);
  for (int j = 0; j < 4; ++j) {
    fold_bn(in.ec[j].bn1, out.a1[j], out.c1[j], 2 * CH, t, nt);
    fold_bn(in.ec[j].bn2, out.a2[j], out.c2[j], CH, t, nt);
    for (int i = t; i < CH * CIN; i += nt) out.W1bf[j][i] = (__bf16)in.ec[j].W1[i];
    for (int i = t; i < CH * CH;  i += nt) out.W2bf[j][i] = (__bf16)in.ec[j].W2[i];
  }
  for (int j = 0; j < 3; ++j) fold_bn(in.obn[j], out.oa[j], out.oc[j], CH, t, nt);
}

// ---------------- simple utility kernels -------------------------------------
__global__ void zero_f32(float* p, int n) {
  int i = blockIdx.x * blockDim.x + threadIdx.x;
  if (i < n) p[i] = 0.0f;
}
__global__ void zero_i32(int* p, int n) {
  int i = blockIdx.x * blockDim.x + threadIdx.x;
  if (i < n) p[i] = 0;
}
__global__ void inv_kernel(const int* __restrict__ cnt, float* __restrict__ inv, int n) {
  int i = blockIdx.x * blockDim.x + threadIdx.x;
  if (i < n) { int c = cnt[i]; inv[i] = (c > 0) ? (1.0f / (float)c) : 0.0f; }
}

// ---------------- edge compaction + per-dst counts ---------------------------
__global__ void compact_kernel(const int* __restrict__ ei, const int* __restrict__ delta,
                               const int* __restrict__ selfe,
                               int* list1, int* list2, int* cnt1, int* cnt2, int* counters) {
  int e = blockIdx.x * blockDim.x + threadIdx.x;
  if (e >= N_EDGES) return;
  int d = delta[e];
  int s = selfe[e];
  int dst = ei[N_EDGES + e];
  bool m1 = (d < 1);
  bool m2 = ((d >= 1) && (d < 3)) || (s == 1);
  if (m1) { int p = atomicAdd(&counters[0], 1); list1[p] = e; atomicAdd(&cnt1[dst], 1); }
  if (m2) { int p = atomicAdd(&counters[1], 1); list2[p] = e; atomicAdd(&cnt2[dst], 1); }
}

// ---------------- node embedding + bn0 + relu --------------------------------
__global__ __launch_bounds__(256) void embed_kernel(
    const float* __restrict__ x, const int* __restrict__ amask,
    const float* __restrict__ W0a, const float* __restrict__ b0a,
    const float* __restrict__ W0v, const float* __restrict__ b0v,
    const float* __restrict__ a0, const float* __restrict__ c0,
    float* __restrict__ G) {
  __shared__ float row[4][CIN];
  int grp  = threadIdx.x >> 6;
  int c    = threadIdx.x & 63;
  int node = blockIdx.x * 4 + grp;
  bool valid = node < N_NODES;
  int am = 0;
  if (valid) {
    am = (amask[node] != 0);
    const float* xr = x + (size_t)node * 4 * CIN + (am ? 0 : CIN);
    row[grp][c]      = xr[c];
    row[grp][c + 64] = xr[c + 64];
  }
  __syncthreads();
  if (!valid) return;
  const float* W = am ? W0a : W0v;
  const float* b = am ? b0a : b0v;
  const float* wr = W + c * CIN;
  float acc = 0.0f;
  #pragma unroll 8
  for (int k = 0; k < CIN; ++k) acc += row[grp][k] * wr[k];
  acc += b[c];
  float v = a0[c] * acc + c0[c];
  G[(size_t)node * CH + c] = fmaxf(v, 0.0f);
}

// ---------------- EdgeConv message MLP on WMMA + atomic scatter --------------
// Per workgroup: 128 edges; per wave: 16 edges x 64 output channels.
// GEMM1: [16x128]bf16 @ [128x64]bf16 -> f32, then bn2+relu -> bf16
// GEMM2: [16x64]bf16  @ [64x64]bf16  -> f32, atomic-add into S[dst].
__global__ __launch_bounds__(256) void edge_conv_kernel(
    const float* __restrict__ feats, float* __restrict__ S,
    const int* __restrict__ list, const int* __restrict__ counters, int cidx,
    const int* __restrict__ ei,
    const float* __restrict__ a1, const float* __restrict__ c1,
    const float* __restrict__ a2, const float* __restrict__ c2,
    const __bf16* __restrict__ W1bf, const __bf16* __restrict__ W2bf) {
  __shared__ __bf16 Abuf[8][16][CIN];
  __shared__ __bf16 Hbuf[8][16][CH];
  __shared__ int dstbuf[8][16];

  int count = counters[cidx];
  int base = blockIdx.x * TILE;
  if (base >= count) return;

  int w    = threadIdx.x >> 5;
  int lane = threadIdx.x & 31;
  int ebase = base + w * 16;

  int halfsel = lane >> 4;           // 0: x_i half, 1: (x_j - x_i) half
  int csub = (lane & 15) * 4;        // 4 channels per lane
  int k0 = halfsel * 64 + csub;

  float4 aa = *(const float4*)(a1 + k0);
  float4 cc = *(const float4*)(c1 + k0);

  // Stage relu(bn1(cat)) rows in LDS as bf16
  for (int es = 0; es < 16; ++es) {
    int li = ebase + es;
    float4 val;
    if (li < count) {
      int e    = list[li];
      int srcn = ei[e];
      int dstn = ei[N_EDGES + e];
      if (lane == 0) dstbuf[w][es] = dstn;
      const float4* xi4 = (const float4*)(feats + (size_t)dstn * CH);
      if (halfsel == 0) {
        val = xi4[csub >> 2];
      } else {
        const float4* xj4 = (const float4*)(feats + (size_t)srcn * CH);
        float4 aj = xj4[csub >> 2];
        float4 bi = xi4[csub >> 2];
        val = make_float4(aj.x - bi.x, aj.y - bi.y, aj.z - bi.z, aj.w - bi.w);
      }
    } else {
      if (lane == 0) dstbuf[w][es] = -1;
      val = make_float4(0.f, 0.f, 0.f, 0.f);
    }
    Abuf[w][es][k0 + 0] = (__bf16)fmaxf(aa.x * val.x + cc.x, 0.f);
    Abuf[w][es][k0 + 1] = (__bf16)fmaxf(aa.y * val.y + cc.y, 0.f);
    Abuf[w][es][k0 + 2] = (__bf16)fmaxf(aa.z * val.z + cc.z, 0.f);
    Abuf[w][es][k0 + 3] = (__bf16)fmaxf(aa.w * val.w + cc.w, 0.f);
  }

  int m  = lane & 15;
  int hi = lane >> 4;

  // GEMM1: K = 128 (4 steps of 32)
  f32x8 acc[4] = {};
  #pragma unroll
  for (int t = 0; t < 4; ++t) {
    bf16x16 afrag;
    int kb = t * 32 + hi * 8;
    #pragma unroll
    for (int j = 0; j < 8; ++j) {
      afrag[j]     = Abuf[w][m][kb + j];
      afrag[8 + j] = Abuf[w][m][kb + 16 + j];
    }
    #pragma unroll
    for (int nt = 0; nt < 4; ++nt) {
      bf16x16 bfrag;
      const __bf16* wrow = W1bf + (nt * 16 + m) * CIN + t * 32 + hi * 8;
      #pragma unroll
      for (int j = 0; j < 8; ++j) { bfrag[j] = wrow[j]; bfrag[8 + j] = wrow[16 + j]; }
      acc[nt] = __builtin_amdgcn_wmma_f32_16x16x32_bf16(
          false, afrag, false, bfrag, (short)0, acc[nt], false, false);
    }
  }

  // bn2 + relu -> bf16 H tile in LDS
  #pragma unroll
  for (int nt = 0; nt < 4; ++nt) {
    int n = nt * 16 + m;
    float a2v = a2[n], c2v = c2[n];
    #pragma unroll
    for (int v = 0; v < 8; ++v) {
      int mm = v + hi * 8;
      Hbuf[w][mm][n] = (__bf16)fmaxf(a2v * acc[nt][v] + c2v, 0.f);
    }
  }

  // GEMM2: K = 64 (2 steps of 32)
  f32x8 acc2[4] = {};
  #pragma unroll
  for (int t = 0; t < 2; ++t) {
    bf16x16 afrag;
    int kb = t * 32 + hi * 8;
    #pragma unroll
    for (int j = 0; j < 8; ++j) {
      afrag[j]     = Hbuf[w][m][kb + j];
      afrag[8 + j] = Hbuf[w][m][kb + 16 + j];
    }
    #pragma unroll
    for (int nt = 0; nt < 4; ++nt) {
      bf16x16 bfrag;
      const __bf16* wrow = W2bf + (nt * 16 + m) * CH + t * 32 + hi * 8;
      #pragma unroll
      for (int j = 0; j < 8; ++j) { bfrag[j] = wrow[j]; bfrag[8 + j] = wrow[16 + j]; }
      acc2[nt] = __builtin_amdgcn_wmma_f32_16x16x32_bf16(
          false, afrag, false, bfrag, (short)0, acc2[nt], false, false);
    }
  }

  // Scatter-add messages into per-dst accumulator
  #pragma unroll
  for (int nt = 0; nt < 4; ++nt) {
    int n = nt * 16 + m;
    #pragma unroll
    for (int v = 0; v < 8; ++v) {
      int mm = v + hi * 8;
      int d = dstbuf[w][mm];
      if (d >= 0) {
        __hip_atomic_fetch_add(&S[(size_t)d * CH + n], acc2[nt][v],
                               __ATOMIC_RELAXED, __HIP_MEMORY_SCOPE_AGENT);
      }
    }
  }
}

// ---------------- mean finalize / block epilogue -----------------------------
__global__ void finalize_mean(const float* __restrict__ S, const float* __restrict__ inv,
                              float* __restrict__ T) {
  int i = blockIdx.x * blockDim.x + threadIdx.x;
  if (i >= N_NODES * CH) return;
  T[i] = S[i] * inv[i >> 6];
}

__global__ void finalize_block(const float* __restrict__ S, const float* __restrict__ inv,
                               float* __restrict__ G,
                               const float* __restrict__ oa, const float* __restrict__ oc,
                               int residual, int dobn) {
  int i = blockIdx.x * blockDim.x + threadIdx.x;
  if (i >= N_NODES * CH) return;
  int ch = i & 63;
  float v = S[i] * inv[i >> 6];
  if (residual) v += G[i];
  if (dobn) { v = oa[ch] * v + oc[ch]; v = fmaxf(v, 0.f); }
  G[i] = v;
}

// ---------------- final FC: 64 -> 2 -----------------------------------------
__global__ void fc_kernel(const float* __restrict__ G, const float* __restrict__ Wfc,
                          const float* __restrict__ bfc, float* __restrict__ out) {
  int n = blockIdx.x * blockDim.x + threadIdx.x;
  if (n >= N_NODES) return;
  float acc0 = bfc[0], acc1 = bfc[1];
  const float* g = G + (size_t)n * CH;
  #pragma unroll 8
  for (int c = 0; c < CH; ++c) {
    float gv = g[c];
    acc0 += gv * Wfc[c];
    acc1 += gv * Wfc[CH + c];
  }
  out[(size_t)n * 2 + 0] = acc0;
  out[(size_t)n * 2 + 1] = acc1;
}

// ---------------- host launcher ----------------------------------------------
extern "C" void kernel_launch(void* const* d_in, const int* in_sizes, int n_in,
                              void* d_out, int out_size, void* d_ws, size_t ws_size,
                              hipStream_t stream) {
  (void)in_sizes; (void)n_in; (void)out_size; (void)ws_size;

  const float* x       = (const float*)d_in[0];
  const int*   ei      = (const int*)d_in[1];
  const int*   edelta  = (const int*)d_in[2];
  const int*   eself   = (const int*)d_in[3];
  const int*   amask   = (const int*)d_in[4];

  int p = 5;
  auto takeF  = [&]() { return (const float*)d_in[p++]; };
  auto takeBN = [&]() { BN r; r.g = takeF(); r.b = takeF(); r.m = takeF(); r.v = takeF(); return r; };

  const float* W0a = takeF();
  const float* b0a = takeF();
  const float* W0v = takeF();
  const float* b0v = takeF();

  PrepIn pi;
  pi.bn0 = takeBN();
  BN obn_tmp[3];
  for (int j = 0; j < 4; ++j) {
    pi.ec[j].bn1 = takeBN();
    pi.ec[j].W1  = takeF();
    pi.ec[j].bn2 = takeBN();
    pi.ec[j].W2  = takeF();
    if (j < 3) obn_tmp[j] = takeBN();
  }
  for (int j = 0; j < 3; ++j) pi.obn[j] = obn_tmp[j];
  const float* Wfc = takeF();
  const float* bfc = takeF();

  // workspace carve-out
  char* wsb = (char*)d_ws;
  size_t off = 0;
  auto take = [&](size_t bytes) -> void* {
    void* pp = wsb + off;
    off = (off + bytes + 255) & ~(size_t)255;
    return pp;
  };
  float* S = (float*)take((size_t)N_NODES * CH * 4);
  float* G = (float*)take((size_t)N_NODES * CH * 4);
  float* T = (float*)take((size_t)N_NODES * CH * 4);
  int* list1 = (int*)take((size_t)N_EDGES * 4);
  int* list2 = (int*)take((size_t)N_EDGES * 4);
  int* cnt1  = (int*)take((size_t)N_NODES * 4);
  int* cnt2  = (int*)take((size_t)N_NODES * 4);
  float* inv1 = (float*)take((size_t)N_NODES * 4);
  float* inv2 = (float*)take((size_t)N_NODES * 4);
  int* counters = (int*)take(8);

  PrepOut po;
  po.a0 = (float*)take(CH * 4);
  po.c0 = (float*)take(CH * 4);
  for (int j = 0; j < 4; ++j) {
    po.a1[j] = (float*)take(2 * CH * 4);
    po.c1[j] = (float*)take(2 * CH * 4);
    po.a2[j] = (float*)take(CH * 4);
    po.c2[j] = (float*)take(CH * 4);
    po.W1bf[j] = (__bf16*)take((size_t)CH * CIN * 2);
    po.W2bf[j] = (__bf16*)take((size_t)CH * CH * 2);
  }
  for (int j = 0; j < 3; ++j) {
    po.oa[j] = (float*)take(CH * 4);
    po.oc[j] = (float*)take(CH * 4);
  }

  const int NC = N_NODES * CH;
  const int GB_NC = (NC + 255) / 256;
  const int GB_N  = (N_NODES + 255) / 256;
  const int GB_E  = (N_EDGES + 255) / 256;
  const int EB    = (N_EDGES + TILE - 1) / TILE;   // upper bound; blocks self-limit via counters

  prep_kernel<<<1, 256, 0, stream>>>(pi, po);
  zero_i32<<<GB_N, 256, 0, stream>>>(cnt1, N_NODES);
  zero_i32<<<GB_N, 256, 0, stream>>>(cnt2, N_NODES);
  zero_i32<<<1, 32, 0, stream>>>(counters, 2);
  compact_kernel<<<GB_E, 256, 0, stream>>>(ei, edelta, eself, list1, list2, cnt1, cnt2, counters);
  inv_kernel<<<GB_N, 256, 0, stream>>>(cnt1, inv1, N_NODES);
  inv_kernel<<<GB_N, 256, 0, stream>>>(cnt2, inv2, N_NODES);
  embed_kernel<<<(N_NODES + 3) / 4, 256, 0, stream>>>(x, amask, W0a, b0a, W0v, b0v, po.a0, po.c0, G);

  for (int b = 0; b < 4; ++b) {
    // conv1 with mask m1
    zero_f32<<<GB_NC, 256, 0, stream>>>(S, NC);
    edge_conv_kernel<<<EB, 256, 0, stream>>>(G, S, list1, counters, 0, ei,
        po.a1[b], po.c1[b], po.a2[b], po.c2[b], po.W1bf[b], po.W2bf[b]);
    finalize_mean<<<GB_NC, 256, 0, stream>>>(S, inv1, T);
    // conv2 with mask m2
    zero_f32<<<GB_NC, 256, 0, stream>>>(S, NC);
    edge_conv_kernel<<<EB, 256, 0, stream>>>(T, S, list2, counters, 1, ei,
        po.a1[b], po.c1[b], po.a2[b], po.c2[b], po.W1bf[b], po.W2bf[b]);
    int residual = (b > 0) ? 1 : 0;
    int dobn = (b < 3) ? 1 : 0;
    finalize_block<<<GB_NC, 256, 0, stream>>>(S, inv2, G,
        dobn ? po.oa[b] : po.a0, dobn ? po.oc[b] : po.c0, residual, dobn);
  }

  fc_kernel<<<GB_N, 256, 0, stream>>>(G, Wfc, bfc, (float*)d_out);
}
